// NestedParallelBlock_57543971832685
// MI455X (gfx1250) — matrix operand order repaired and checked by reading, not compiled, not run on hardware
//
#include <hip/hip_runtime.h>
#include <hip/hip_bf16.h>
#include <math.h>

// ---------------- types ----------------
typedef __attribute__((ext_vector_type(16))) __bf16 bf16x16;
typedef __attribute__((ext_vector_type(8)))  __bf16 bf16x8;
typedef __attribute__((ext_vector_type(8)))  float  f32x8;
typedef int v4i_gcc __attribute__((vector_size(16)));   // matches builtin's V4i

#define DIM_      1024
#define TOKENS_   8192      // B*N = 8*1024
#define EXPAND_   7168      // 3*DIM + 4*DIM
#define HID_      4096
#define CATIN_    5120      // 4*DIM + DIM
#define CONOUT_   2048
#define HEADS_    16
#define DH_       64

// ---------------- CDNA5 async global->LDS path (guarded; falls back cleanly) ----
#if defined(__gfx1250__) && __has_builtin(__builtin_amdgcn_global_load_async_to_lds_b128)
#define HAS_ASYNC_LDS 1
#else
#define HAS_ASYNC_LDS 0
#endif

#define AS1 __attribute__((address_space(1)))
#define AS3 __attribute__((address_space(3)))

__device__ __forceinline__ void async_b128(void* lds, const void* g) {
#if HAS_ASYNC_LDS
  __builtin_amdgcn_global_load_async_to_lds_b128(
      (AS1 v4i_gcc*)g, (AS3 v4i_gcc*)lds, 0, 0);
#else
  (void)lds; (void)g;
#endif
}

#if HAS_ASYNC_LDS
#if __has_builtin(__builtin_amdgcn_s_wait_asynccnt)
#define WAIT_ASYNC(n) __builtin_amdgcn_s_wait_asynccnt(n)
#else
#define WAIT_ASYNC(n) asm volatile("s_wait_asynccnt %0" :: "i"(n) : "memory")
#endif
#define GEMM_BUFS 2
#else
#define WAIT_ASYNC(n)
#define GEMM_BUFS 1
#endif

__device__ __forceinline__ unsigned short f2bf(float f) {
  unsigned int u = __float_as_uint(f);
  u += 0x7FFFu + ((u >> 16) & 1u);     // round-to-nearest-even
  return (unsigned short)(u >> 16);
}

__device__ __forceinline__ bf16x16 cat8(bf16x8 lo, bf16x8 hi) {
  return __builtin_shufflevector(lo, hi, 0,1,2,3,4,5,6,7,8,9,10,11,12,13,14,15);
}

// ---------------- fp32 -> bf16 conversion ----------------
__global__ __launch_bounds__(256) void conv_kernel(const float* __restrict__ in,
                                                   unsigned short* __restrict__ out, int n) {
  int i = blockIdx.x * 256 + threadIdx.x;
  if (i < n) out[i] = f2bf(in[i]);
}

// ---------------- block reduction helper ----------------
__device__ __forceinline__ void block_reduce2(float& s, float& s2) {
  #pragma unroll
  for (int off = 16; off > 0; off >>= 1) {
    s  += __shfl_xor(s,  off, 32);
    s2 += __shfl_xor(s2, off, 32);
  }
  __shared__ float rs[8], rs2[8];
  int wid = threadIdx.x >> 5;
  if ((threadIdx.x & 31) == 0) { rs[wid] = s; rs2[wid] = s2; }
  __syncthreads();
  float a = 0.f, b = 0.f;
  #pragma unroll
  for (int i = 0; i < 8; ++i) { a += rs[i]; b += rs2[i]; }
  s = a; s2 = b;
  __syncthreads();
}

// ---------------- LN1 + input channel mask -> bf16 ----------------
__global__ __launch_bounds__(256) void ln1_kernel(const float* __restrict__ x,
                                                  const float* __restrict__ w,
                                                  const float* __restrict__ bb,
                                                  const int*   __restrict__ emask,
                                                  unsigned short* __restrict__ xn) {
  int tok = blockIdx.x;
  const float* xr = x + (size_t)tok * DIM_;
  float v[4]; float s = 0.f, s2 = 0.f;
  #pragma unroll
  for (int i = 0; i < 4; ++i) {
    v[i] = xr[threadIdx.x + i * 256];
    s += v[i]; s2 += v[i] * v[i];
  }
  block_reduce2(s, s2);
  float mu  = s  * (1.f / DIM_);
  float var = s2 * (1.f / DIM_) - mu * mu;
  float rq  = rsqrtf(var + 1e-5f);
  int d_in = 128 << emask[tok];          // 1024 >> (3 - mask)
  #pragma unroll
  for (int i = 0; i < 4; ++i) {
    int c = threadIdx.x + i * 256;
    float val = (v[i] - mu) * rq * w[c] + bb[c];
    if (c >= d_in) val = 0.f;
    xn[(size_t)tok * DIM_ + c] = f2bf(val);
  }
}

// ---------------- LN2 over kv (2048) -> k_bf, v_bf ----------------
__global__ __launch_bounds__(256) void ln2_kernel(const float* __restrict__ kv,
                                                  const float* __restrict__ w,
                                                  const float* __restrict__ bb,
                                                  unsigned short* __restrict__ k_bf,
                                                  unsigned short* __restrict__ v_bf) {
  int tok = blockIdx.x;
  const float* r = kv + (size_t)tok * 2048;
  float v[8]; float s = 0.f, s2 = 0.f;
  #pragma unroll
  for (int i = 0; i < 8; ++i) {
    v[i] = r[threadIdx.x + i * 256];
    s += v[i]; s2 += v[i] * v[i];
  }
  block_reduce2(s, s2);
  float mu  = s  * (1.f / 2048.f);
  float var = s2 * (1.f / 2048.f) - mu * mu;
  float rq  = rsqrtf(var + 1e-5f);
  #pragma unroll
  for (int i = 0; i < 8; ++i) {
    int c = threadIdx.x + i * 256;
    float val = (v[i] - mu) * rq * w[c] + bb[c];
    unsigned short h = f2bf(val);
    if (c < 1024) k_bf[(size_t)tok * DIM_ + c] = h;
    else          v_bf[(size_t)tok * DIM_ + (c - 1024)] = h;
  }
}

// ---------------- shared WMMA GEMM mainloop: C[128x128] += A(MxK) * B(NxK)^T ----------------
// 8 waves: wave grid 4(M) x 2(N), each wave 32x64 = 2x4 wmma tiles.
// Async path: double-buffered LDS via GLOBAL_LOAD_ASYNC_TO_LDS_B128 + s_wait_asynccnt.
__device__ __forceinline__ void gemm_tile_bf16(const unsigned short* __restrict__ A,
                                               const unsigned short* __restrict__ Bw,
                                               int K, int m0, int n0,
                                               unsigned short* sA, unsigned short* sB,
                                               f32x8 (&acc)[2][4]) {
  const int t = threadIdx.x;
  const int lane = t & 31, wid = t >> 5;
  const int wm = wid >> 1, wn = wid & 1;
  const int rA = t >> 1;
  const int cA = (t & 1) * 16;

  #pragma unroll
  for (int mt = 0; mt < 2; ++mt)
    #pragma unroll
    for (int nt = 0; nt < 4; ++nt)
      #pragma unroll
      for (int j = 0; j < 8; ++j) acc[mt][nt][j] = 0.f;

  const int kTiles = K >> 5;

#if HAS_ASYNC_LDS
  // issue tile kt's copies into buffer buf (4 x b128 per thread)
  auto issue = [&](int buf, int kt) {
    const int k0 = kt << 5;
    const unsigned short* ga = A  + (size_t)(m0 + rA) * K + k0 + cA;
    const unsigned short* gb = Bw + (size_t)(n0 + rA) * K + k0 + cA;
    unsigned short* la = sA + buf * (128 * 32) + rA * 32 + cA;
    unsigned short* lb = sB + buf * (128 * 32) + rA * 32 + cA;
    async_b128(la,     ga);
    async_b128(la + 8, ga + 8);
    async_b128(lb,     gb);
    async_b128(lb + 8, gb + 8);
  };
  issue(0, 0);
  for (int kt = 0; kt < kTiles; ++kt) {
    const int cur = kt & 1;
    if (kt + 1 < kTiles) {         // prefetch next tile into other buffer
      issue(cur ^ 1, kt + 1);
      WAIT_ASYNC(4);               // current tile's 4 copies complete (in-order)
    } else {
      WAIT_ASYNC(0);
    }
    __syncthreads();

    const unsigned short* sAc = sA + cur * (128 * 32);
    const unsigned short* sBc = sB + cur * (128 * 32);
    bf16x16 aF[2], bF[4];
    #pragma unroll
    for (int mt = 0; mt < 2; ++mt) {
      int row = wm * 32 + mt * 16 + (lane & 15);
      int kb  = (lane >> 4) << 4;
      bf16x8 lo = *reinterpret_cast<const bf16x8*>(sAc + row * 32 + kb);
      bf16x8 hi = *reinterpret_cast<const bf16x8*>(sAc + row * 32 + kb + 8);
      aF[mt] = cat8(lo, hi);
    }
    #pragma unroll
    for (int nt = 0; nt < 4; ++nt) {
      int row = wn * 64 + nt * 16 + (lane & 15);
      int kb  = (lane >> 4) << 4;
      bF[nt] = *reinterpret_cast<const bf16x16*>(sBc + row * 32 + kb);
    }
    #pragma unroll
    for (int mt = 0; mt < 2; ++mt)
      #pragma unroll
      for (int nt = 0; nt < 4; ++nt)
        acc[mt][nt] = __builtin_amdgcn_wmma_f32_16x16x32_bf16(
            false, aF[mt], false, bF[nt], (short)0, acc[mt][nt], false, false);
    __syncthreads();
  }
#else
  for (int kt = 0; kt < kTiles; ++kt) {
    const int k0 = kt << 5;
    const uint4* ga = reinterpret_cast<const uint4*>(A  + (size_t)(m0 + rA) * K + k0 + cA);
    const uint4* gb = reinterpret_cast<const uint4*>(Bw + (size_t)(n0 + rA) * K + k0 + cA);
    if (kt + 1 < kTiles) {                    // hint next K tile into caches
      __builtin_prefetch(reinterpret_cast<const char*>(ga) + 64, 0, 3);
      __builtin_prefetch(reinterpret_cast<const char*>(gb) + 64, 0, 3);
    }
    uint4 a0 = ga[0], a1 = ga[1];
    uint4 b0 = gb[0], b1 = gb[1];
    *reinterpret_cast<uint4*>(sA + rA * 32 + cA)     = a0;
    *reinterpret_cast<uint4*>(sA + rA * 32 + cA + 8) = a1;
    *reinterpret_cast<uint4*>(sB + rA * 32 + cA)     = b0;
    *reinterpret_cast<uint4*>(sB + rA * 32 + cA + 8) = b1;
    __syncthreads();

    bf16x16 aF[2], bF[4];
    #pragma unroll
    for (int mt = 0; mt < 2; ++mt) {
      int row = wm * 32 + mt * 16 + (lane & 15);
      int kb  = (lane >> 4) << 4;
      bf16x8 lo = *reinterpret_cast<const bf16x8*>(sA + row * 32 + kb);
      bf16x8 hi = *reinterpret_cast<const bf16x8*>(sA + row * 32 + kb + 8);
      aF[mt] = cat8(lo, hi);
    }
    #pragma unroll
    for (int nt = 0; nt < 4; ++nt) {
      int row = wn * 64 + nt * 16 + (lane & 15);
      int kb  = (lane >> 4) << 4;
      bF[nt] = *reinterpret_cast<const bf16x16*>(sB + row * 32 + kb);
    }
    #pragma unroll
    for (int mt = 0; mt < 2; ++mt)
      #pragma unroll
      for (int nt = 0; nt < 4; ++nt)
        acc[mt][nt] = __builtin_amdgcn_wmma_f32_16x16x32_bf16(
            false, aF[mt], false, bF[nt], (short)0, acc[mt][nt], false, false);
    __syncthreads();
  }
#endif
}

// ---------------- GEMM1: y = xn * Wexp^T (+ input_bias), fused split epilogue ----------------
__global__ __launch_bounds__(256) void gemm1_kernel(const unsigned short* __restrict__ xn,
                                                    const unsigned short* __restrict__ Wexp,
                                                    const float* __restrict__ mlp_bias,
                                                    unsigned short* __restrict__ q_bf,
                                                    float* __restrict__ kv_f32,
                                                    unsigned short* __restrict__ cat_bf) {
  __shared__ unsigned short sA[GEMM_BUFS * 128 * 32];
  __shared__ unsigned short sB[GEMM_BUFS * 128 * 32];
  f32x8 acc[2][4];
  const int m0 = blockIdx.y * 128, n0 = blockIdx.x * 128;
  gemm_tile_bf16(xn, Wexp, DIM_, m0, n0, sA, sB, acc);

  const int lane = threadIdx.x & 31, wid = threadIdx.x >> 5;
  const int wm = wid >> 1, wn = wid & 1;
  const int rbase = m0 + wm * 32 + ((lane >> 4) << 3);
  const int cbase = n0 + wn * 64 + (lane & 15);

  // Region boundaries (1024/3072/4096) are multiples of 128 -> block-uniform branch.
  if (n0 < DIM_) {                       // q region (input_bias = mlp_bias[col] here)
    #pragma unroll
    for (int mt = 0; mt < 2; ++mt)
      #pragma unroll
      for (int nt = 0; nt < 4; ++nt)
        #pragma unroll
        for (int g = 0; g < 8; ++g) {
          int row = rbase + mt * 16 + g;
          int col = cbase + nt * 16;
          q_bf[(size_t)row * DIM_ + col] = f2bf(acc[mt][nt][g] + mlp_bias[col]);
        }
  } else if (n0 < 3 * DIM_) {            // kv region -> fp32 for LN2
    #pragma unroll
    for (int mt = 0; mt < 2; ++mt)
      #pragma unroll
      for (int nt = 0; nt < 4; ++nt)
        #pragma unroll
        for (int g = 0; g < 8; ++g) {
          int row = rbase + mt * 16 + g;
          int col = cbase + nt * 16;
          kv_f32[(size_t)row * 2048 + (col - DIM_)] = acc[mt][nt][g] + mlp_bias[col];
        }
  } else {                               // mlp hidden -> exact gelu -> concat buffer
    const bool hasb = (n0 < HID_);       // bias stops at col 4096 (block-uniform)
    #pragma unroll
    for (int mt = 0; mt < 2; ++mt)
      #pragma unroll
      for (int nt = 0; nt < 4; ++nt)
        #pragma unroll
        for (int g = 0; g < 8; ++g) {
          int row = rbase + mt * 16 + g;
          int col = cbase + nt * 16;
          float v = acc[mt][nt][g];
          if (hasb) v += mlp_bias[col];
          float gg = 0.5f * v * (1.f + erff(v * 0.70710678118654752f));
          cat_bf[(size_t)row * CATIN_ + (col - 3 * DIM_)] = f2bf(gg);
        }
  }
}

// ---------------- GEMM2: cy = concat * Wcon^T + bias ----------------
__global__ __launch_bounds__(256) void gemm2_kernel(const unsigned short* __restrict__ cat_bf,
                                                    const unsigned short* __restrict__ Wcon,
                                                    const float* __restrict__ cbias,
                                                    float* __restrict__ cy) {
  __shared__ unsigned short sA[GEMM_BUFS * 128 * 32];
  __shared__ unsigned short sB[GEMM_BUFS * 128 * 32];
  f32x8 acc[2][4];
  const int m0 = blockIdx.y * 128, n0 = blockIdx.x * 128;
  gemm_tile_bf16(cat_bf, Wcon, CATIN_, m0, n0, sA, sB, acc);

  const int lane = threadIdx.x & 31, wid = threadIdx.x >> 5;
  const int wm = wid >> 1, wn = wid & 1;
  #pragma unroll
  for (int mt = 0; mt < 2; ++mt)
    #pragma unroll
    for (int nt = 0; nt < 4; ++nt)
      #pragma unroll
      for (int g = 0; g < 8; ++g) {
        int row = m0 + wm * 32 + mt * 16 + ((lane >> 4) << 3) + g;
        int col = n0 + wn * 64 + nt * 16 + (lane & 15);
        cy[(size_t)row * CONOUT_ + col] = acc[mt][nt][g] + cbias[col];
      }
}

// ---------------- flash attention: per (b, h, 128-query tile) ----------------
__global__ __launch_bounds__(256) void attn_kernel(const unsigned short* __restrict__ q_bf,
                                                   const unsigned short* __restrict__ k_bf,
                                                   const unsigned short* __restrict__ v_bf,
                                                   unsigned short* __restrict__ cat_bf) {
  __shared__ unsigned short sQ[128 * 64];
  __shared__ unsigned short sK[32 * 64];
  __shared__ unsigned short sVt[64 * 32];
  __shared__ unsigned short sP[8 * 16 * 32];

  const int bx = blockIdx.x;
  const int qt = bx & 7;
  const int h  = (bx >> 3) & 15;
  const int b  = bx >> 7;
  const int t = threadIdx.x, lane = t & 31, wid = t >> 5;

  const size_t qbase = ((size_t)b * 1024 + (size_t)qt * 128) * DIM_ + h * DH_;
  #pragma unroll
  for (int u = 0; u < 4; ++u) {
    int id = t + u * 256;              // 1024 x b128 (128 rows * 8)
    int r = id >> 3, c = (id & 7) * 8;
#if HAS_ASYNC_LDS
    async_b128(sQ + r * 64 + c, q_bf + qbase + (size_t)r * DIM_ + c);
#else
    *reinterpret_cast<uint4*>(sQ + r * 64 + c) =
        *reinterpret_cast<const uint4*>(q_bf + qbase + (size_t)r * DIM_ + c);
#endif
  }
  WAIT_ASYNC(0);
  __syncthreads();

  // Q fragments (hoisted): 16 query rows per wave, contract dh=64 -> 2 K-steps
  bf16x16 aQ[2];
  {
    int row = wid * 16 + (lane & 15);
    #pragma unroll
    for (int ks = 0; ks < 2; ++ks) {
      int kb = ks * 32 + ((lane >> 4) << 4);
      bf16x8 lo = *reinterpret_cast<const bf16x8*>(sQ + row * 64 + kb);
      bf16x8 hi = *reinterpret_cast<const bf16x8*>(sQ + row * 64 + kb + 8);
      aQ[ks] = cat8(lo, hi);
    }
  }

  f32x8 Oacc[4];
  float mrow[8], lrow[8];
  #pragma unroll
  for (int nt = 0; nt < 4; ++nt)
    #pragma unroll
    for (int j = 0; j < 8; ++j) Oacc[nt][j] = 0.f;
  #pragma unroll
  for (int g = 0; g < 8; ++g) { mrow[g] = -3.0e38f; lrow[g] = 0.f; }

  const float scale = 0.125f;           // dh^-0.5, dh = 64

  for (int kk = 0; kk < 32; ++kk) {     // 32 key chunks of 32
    const size_t kvbase = ((size_t)b * 1024 + (size_t)kk * 32) * DIM_ + h * DH_;
    {
      int r = t >> 3, c = (t & 7) * 8;  // 256 x b128 = 32x64 tile
#if HAS_ASYNC_LDS
      async_b128(sK + r * 64 + c, k_bf + kvbase + (size_t)r * DIM_ + c);
#else
      *reinterpret_cast<uint4*>(sK + r * 64 + c) =
          *reinterpret_cast<const uint4*>(k_bf + kvbase + (size_t)r * DIM_ + c);
#endif
    }
    #pragma unroll
    for (int u = 0; u < 8; ++u) {       // V transposed: sVt[dh][key]
      int id = t + u * 256;             // 0..2047
      int key = id >> 6, dc = id & 63;
      sVt[dc * 32 + key] = v_bf[kvbase + (size_t)key * DIM_ + dc];
    }
    WAIT_ASYNC(0);
    __syncthreads();

    // S = Q * K^T  (16 q-rows x 32 keys, two 16x16 tiles)
    f32x8 S[2];
    #pragma unroll
    for (int st = 0; st < 2; ++st) {
      #pragma unroll
      for (int j = 0; j < 8; ++j) S[st][j] = 0.f;
      #pragma unroll
      for (int ks = 0; ks < 2; ++ks) {
        int row = st * 16 + (lane & 15);
        int kb  = ks * 32 + ((lane >> 4) << 4);
        bf16x16 bK = *reinterpret_cast<const bf16x16*>(sK + row * 64 + kb);
        S[st] = __builtin_amdgcn_wmma_f32_16x16x32_bf16(
            false, aQ[ks], false, bK, (short)0, S[st], false, false);
      }
    }

    // online softmax (row m = 8*(lane>=16)+g lives in one 16-lane half)
    float pr[2][8];
    #pragma unroll
    for (int g = 0; g < 8; ++g) {
      float s0 = S[0][g] * scale;
      float s1 = S[1][g] * scale;
      float mx = fmaxf(s0, s1);
      #pragma unroll
      for (int off = 1; off < 16; off <<= 1) mx = fmaxf(mx, __shfl_xor(mx, off, 32));
      float mnew  = fmaxf(mrow[g], mx);
      float alpha = __expf(mrow[g] - mnew);
      float p0 = __expf(s0 - mnew);
      float p1 = __expf(s1 - mnew);
      float ls = p0 + p1;
      #pragma unroll
      for (int off = 1; off < 16; off <<= 1) ls += __shfl_xor(ls, off, 32);
      lrow[g] = lrow[g] * alpha + ls;
      mrow[g] = mnew;
      pr[0][g] = p0; pr[1][g] = p1;
      #pragma unroll
      for (int nt = 0; nt < 4; ++nt) Oacc[nt][g] *= alpha;
    }

    // C-layout -> row-major P in per-wave LDS scratch, reload as A fragment
    unsigned short* sPw = sP + wid * 512;
    #pragma unroll
    for (int st = 0; st < 2; ++st)
      #pragma unroll
      for (int g = 0; g < 8; ++g) {
        int m = ((lane >> 4) << 3) + g;
        int n = st * 16 + (lane & 15);
        sPw[m * 32 + n] = f2bf(pr[st][g]);
      }
    bf16x16 aP;
    {
      int row = lane & 15;
      int kb  = (lane >> 4) << 4;
      bf16x8 lo = *reinterpret_cast<const bf16x8*>(sPw + row * 32 + kb);
      bf16x8 hi = *reinterpret_cast<const bf16x8*>(sPw + row * 32 + kb + 8);
      aP = cat8(lo, hi);
    }
    #pragma unroll
    for (int nt = 0; nt < 4; ++nt) {
      int row = nt * 16 + (lane & 15);
      int kb  = (lane >> 4) << 4;
      bf16x16 bV = *reinterpret_cast<const bf16x16*>(sVt + row * 32 + kb);
      Oacc[nt] = __builtin_amdgcn_wmma_f32_16x16x32_bf16(
          false, aP, false, bV, (short)0, Oacc[nt], false, false);
    }
    __syncthreads();
  }

  // write normalized output into concat buffer cols [4096 + h*64 ...)
  #pragma unroll
  for (int nt = 0; nt < 4; ++nt)
    #pragma unroll
    for (int g = 0; g < 8; ++g) {
      int m = ((lane >> 4) << 3) + g;
      size_t row = (size_t)b * 1024 + qt * 128 + wid * 16 + m;
      int col = HID_ + h * DH_ + nt * 16 + (lane & 15);
      cat_bf[row * CATIN_ + col] = f2bf(Oacc[nt][g] / lrow[g]);
    }
}

// ---------------- final: mask + residual + expert-prob combine ----------------
__global__ __launch_bounds__(256) void final_kernel(const float* __restrict__ cy,
                                                    const float* __restrict__ x,
                                                    const float* __restrict__ probs,
                                                    const int*   __restrict__ emask,
                                                    float* __restrict__ out) {
  int idx = blockIdx.x * 256 + threadIdx.x;
  int tok = idx >> 10, c = idx & 1023;
  int d_out = 256 << emask[tok];         // 2048 >> (3 - mask)
  float xm = x[idx];
  float cm = cy[(size_t)tok * CONOUT_ + c];
  float ca = cy[(size_t)tok * CONOUT_ + 1024 + c];
  float mlp = (c < d_out ? cm : 0.f) + xm;
  float att = ((1024 + c) < d_out ? ca : 0.f) + xm;
  out[idx] = att + probs[tok] * mlp;
}

// ---------------- launcher ----------------
extern "C" void kernel_launch(void* const* d_in, const int* in_sizes, int n_in,
                              void* d_out, int out_size, void* d_ws, size_t ws_size,
                              hipStream_t stream) {
  (void)in_sizes; (void)n_in; (void)out_size; (void)ws_size;
  const float* x        = (const float*)d_in[0];
  const float* probs    = (const float*)d_in[1];
  const float* Wexp     = (const float*)d_in[2];
  const float* mlp_bias = (const float*)d_in[3];
  const float* Wcon     = (const float*)d_in[4];
  const float* cbias    = (const float*)d_in[5];
  const float* n1w      = (const float*)d_in[6];
  const float* n1b      = (const float*)d_in[7];
  const float* n2w      = (const float*)d_in[8];
  const float* n2b      = (const float*)d_in[9];
  const int*   emask    = (const int*)d_in[10];
  float* out = (float*)d_out;

  char* ws = (char*)d_ws;
  size_t off = 0;
  auto alloc = [&](size_t bytes) -> void* {
    void* p = ws + off;
    off = (off + bytes + 255) & ~(size_t)255;
    return p;
  };
  unsigned short* Wexp_bf = (unsigned short*)alloc((size_t)EXPAND_ * DIM_ * 2);
  unsigned short* Wcon_bf = (unsigned short*)alloc((size_t)CONOUT_ * CATIN_ * 2);
  unsigned short* xn_bf   = (unsigned short*)alloc((size_t)TOKENS_ * DIM_ * 2);
  unsigned short* q_bf    = (unsigned short*)alloc((size_t)TOKENS_ * DIM_ * 2);
  float*          kv_cy   = (float*)alloc((size_t)TOKENS_ * 2048 * 4);  // kv fp32, later reused as cy
  unsigned short* k_bf    = (unsigned short*)alloc((size_t)TOKENS_ * DIM_ * 2);
  unsigned short* v_bf    = (unsigned short*)alloc((size_t)TOKENS_ * DIM_ * 2);
  unsigned short* cat_bf  = (unsigned short*)alloc((size_t)TOKENS_ * CATIN_ * 2);

  const int nWexp = EXPAND_ * DIM_;
  const int nWcon = CONOUT_ * CATIN_;
  conv_kernel<<<(nWexp + 255) / 256, 256, 0, stream>>>(Wexp, Wexp_bf, nWexp);
  conv_kernel<<<(nWcon + 255) / 256, 256, 0, stream>>>(Wcon, Wcon_bf, nWcon);

  ln1_kernel<<<TOKENS_, 256, 0, stream>>>(x, n1w, n1b, emask, xn_bf);

  gemm1_kernel<<<dim3(EXPAND_ / 128, TOKENS_ / 128), 256, 0, stream>>>(
      xn_bf, Wexp_bf, mlp_bias, q_bf, kv_cy, cat_bf);

  ln2_kernel<<<TOKENS_, 256, 0, stream>>>(kv_cy, n2w, n2b, k_bf, v_bf);

  attn_kernel<<<8 * HEADS_ * 8, 256, 0, stream>>>(q_bf, k_bf, v_bf, cat_bf);

  gemm2_kernel<<<dim3(CONOUT_ / 128, TOKENS_ / 128), 256, 0, stream>>>(
      cat_bf, Wcon_bf, cbias, kv_cy);

  final_kernel<<<(TOKENS_ * DIM_) / 256, 256, 0, stream>>>(kv_cy, x, probs, emask, out);
}
